// GRUEncoder_6502580486599
// MI455X (gfx1250) — matrix operand order repaired and checked by reading, not compiled, not run on hardware
//
#include <hip/hip_runtime.h>
#include <hip/hip_bf16.h>

typedef __attribute__((ext_vector_type(16))) _Float16 v16h;
typedef __attribute__((ext_vector_type(8)))  _Float16 v8h;
typedef __attribute__((ext_vector_type(8)))  float    v8f;

#define H       256
#define G3      768          // 3*H
#define SEQ     1024
#define BATCH   64
#define NROWS   (BATCH*SEQ)  // 65536
#define LDS_STRIDE 264       // 256 + 8 halves padding -> conflict-free b128 A loads

// swizzled weight block for one gate: 16 tiles x 8 kc x 32 lanes x 16 halves
#define GATE_HALVES (16 * 8 * 32 * 16)          // 65536 halves = 128 KB
#define LDSB_HALVES (2 * GATE_HALVES)           // gates r,z resident: 256 KB
#define HBUF_HALVES (16 * LDS_STRIDE)           // one h buffer: 8448 B

__device__ __forceinline__ float sigmoid_fast(float x) {
    return 1.0f / (1.0f + __expf(-x));
}
__device__ __forceinline__ float tanh_fast(float x) {
    // 1 - 2/(e^{2x}+1): saturates correctly for |x| large, no inf/inf
    float e = __expf(2.0f * x);
    return 1.0f - 2.0f / (e + 1.0f);
}

__device__ __forceinline__ v8f wmma_f16(v16h a, v16h b, v8f c) {
    return __builtin_amdgcn_wmma_f32_16x16x32_f16(
        /*neg_a=*/false, a, /*neg_b=*/false, b,
        /*c_mod=*/(short)0, c, /*reuse_a=*/false, /*reuse_b=*/false);
}

// Build a v16h A-operand from LDS: lane m = lane&15, K-halves [c0..c0+7] and [c0+16..c0+23]
__device__ __forceinline__ v16h load_A(const _Float16* base, int mrow, int kc, int hoff) {
    const _Float16* ap = base + mrow * LDS_STRIDE + kc * 32 + hoff;
    v8h a0 = *(const v8h*)(ap);
    v8h a1 = *(const v8h*)(ap + 16);
    return __builtin_shufflevector(a0, a1,
        0,1,2,3,4,5,6,7,8,9,10,11,12,13,14,15);
}

// ---------------------------------------------------------------------------
// Kernel 1: convert a 768x256 f32 weight matrix into the per-lane swizzled
// f16 B-matrix layout:
//   dst[((nt*8 + kc)*32 + lane)*16 + hh] = W[nt*16 + (lane&15)][kc*32 + 8*(lane>>4) + hh + 8*(hh>>3)]
// so the GEMM kernels fetch one tile chunk with one contiguous 32-byte (v16h) load per lane.
// ---------------------------------------------------------------------------
__global__ __launch_bounds__(256) void pack_weights(const float* __restrict__ W,
                                                    unsigned short* __restrict__ dst_u16) {
    int idx = blockIdx.x * 256 + threadIdx.x;      // < 768*256 = 196608
    int hh = idx & 15;
    int l  = (idx >> 4) & 31;
    int kc = (idx >> 9) & 7;
    int nt = idx >> 12;
    int n = nt * 16 + (l & 15);
    int k = kc * 32 + ((l >> 4) << 3) + hh + ((hh >> 3) << 3);
    _Float16* dst = (_Float16*)dst_u16;
    dst[idx] = (_Float16)W[n * H + k];
}

// ---------------------------------------------------------------------------
// Kernel 2: fused embedding gather + input projection (WMMA f16, f32 accum).
// One block = one 16-row tile of the [65536, 768] output; 8 waves x 6 N-tiles.
// ---------------------------------------------------------------------------
__global__ __launch_bounds__(256) void xproj_kernel(const int* __restrict__ inputs,
                                                    const float* __restrict__ emb,
                                                    const unsigned short* __restrict__ wih_swz_u,
                                                    const float* __restrict__ b_ih,
                                                    unsigned short* __restrict__ xproj_u) {
    __shared__ __align__(32) unsigned short aL_u[16 * LDS_STRIDE];
    _Float16* aL = (_Float16*)aL_u;

    const int tid = threadIdx.x;
    const int rt  = blockIdx.x;            // row tile: rows [rt*16, rt*16+16)

    // Gather 16 embedding rows (f32 -> f16) into LDS
    {
        int m   = tid >> 4;                // 0..15
        int seg = tid & 15;                // 16 floats each
        int row = rt * 16 + m;
        int tok = inputs[row];
        const float* src = emb + (size_t)tok * H + seg * 16;
        _Float16* d = aL + m * LDS_STRIDE + seg * 16;
        #pragma unroll
        for (int i = 0; i < 16; ++i) d[i] = (_Float16)src[i];
    }
    __syncthreads();

    const int w    = tid >> 5;             // wave 0..7
    const int lane = tid & 31;
    const int mrow = lane & 15;
    const int hoff = (lane >> 4) << 3;     // 0 or 8
    const _Float16* wih = (const _Float16*)wih_swz_u;
    _Float16* xp = (_Float16*)xproj_u;

    for (int ti = 0; ti < 6; ++ti) {
        int nt = w * 6 + ti;               // N-tile 0..47
        v8f acc = {};
        #pragma unroll
        for (int kc = 0; kc < 8; ++kc) {
            v16h a = load_A(aL, mrow, kc, hoff);
            v16h b = *(const v16h*)(wih + (((size_t)nt * 8 + kc) * 32 + lane) * 16);
            acc = wmma_f16(a, b, acc);
        }
        int ncol = nt * 16 + (lane & 15);
        float bias = b_ih[ncol];
        #pragma unroll
        for (int j = 0; j < 8; ++j) {
            int r = rt * 16 + j + hoff;    // C layout: M = j (+8 for upper lanes)
            xp[(size_t)r * G3 + ncol] = (_Float16)(acc[j] + bias);
        }
    }
}

// ---------------------------------------------------------------------------
// Kernel 3: sequential GRU scan. 4 blocks x 512 threads; block bt owns batches
// [bt*16, bt*16+16). 16 waves: wave w owns hidden columns [w*16, w*16+16) and
// computes the r/z/n projections for them (3 WMMA accumulators, K=256).
// W_hh gates r,z pinned in LDS (256 KB, staged once); n-gate streamed from L2.
// h double-buffered in LDS (f16).  Total LDS: 279 KB of the WGP's 320 KB.
// ---------------------------------------------------------------------------
__global__ __launch_bounds__(512) void scan_kernel(const unsigned short* __restrict__ whh_swz_u,
                                                   const unsigned short* __restrict__ xproj_u,
                                                   const float* __restrict__ b_hh,
                                                   float* __restrict__ out) {
    __shared__ __align__(32) unsigned short smem[LDSB_HALVES + 2 * HBUF_HALVES];

    const int tid = threadIdx.x;

    // Stage gates r,z of swizzled W_hh into LDS (first 2*GATE_HALVES halves are
    // exactly tiles nt=0..31, i.e. gates 0 and 1, in the flat swizzled order).
    {
        const uint4* src = (const uint4*)whh_swz_u;       // 256 KB = 16384 uint4
        uint4* dst = (uint4*)smem;
        for (int i = tid; i < (LDSB_HALVES * 2) / 16; i += 512) dst[i] = src[i];
    }
    // h0 = 0 (f16 zero bit pattern is 0)
    for (int i = tid; i < 2 * HBUF_HALVES; i += 512) smem[LDSB_HALVES + i] = 0;
    __syncthreads();

    const int w    = tid >> 5;             // wave 0..15 -> column tile
    const int lane = tid & 31;
    const int mrow = lane & 15;
    const int hoff = (lane >> 4) << 3;     // 0 or 8 (A K-offset AND C M-offset)
    const int bt   = blockIdx.x;           // batch tile 0..3
    const int col  = w * 16 + (lane & 15); // hidden column 0..255

    const _Float16* xp = (const _Float16*)xproj_u;
    float* ann = out;                                 // [B,S,H]
    float* hid = out + (size_t)BATCH * SEQ * H;       // [B,H]

    const float bh_r = b_hh[col];
    const float bh_z = b_hh[H + col];
    const float bh_n = b_hh[2 * H + col];

    // LDS-resident B bases for this wave's column tile (gates r,z)
    const _Float16* bLr = (const _Float16*)smem + ((size_t)(0 * 16 + w) * 8) * 32 * 16;
    const _Float16* bLz = (const _Float16*)smem + ((size_t)(1 * 16 + w) * 8) * 32 * 16;
    // n-gate stays in global (L2-hot, 128 KB/step)
    const _Float16* wbn = (const _Float16*)whh_swz_u + ((size_t)(2 * 16 + w) * 8) * 32 * 16;

    _Float16* hbuf = (_Float16*)(smem + LDSB_HALVES);

    for (int t = 0; t < SEQ; ++t) {
        const int p = t & 1, q = p ^ 1;
        const _Float16* hcur = hbuf + p * HBUF_HALVES;
        _Float16*       hnxt = hbuf + q * HBUF_HALVES;

        v8f ar = {}, az = {}, an = {};
        #pragma unroll
        for (int kc = 0; kc < 8; ++kc) {
            v16h a = load_A(hcur, mrow, kc, hoff);
            size_t co = ((size_t)kc * 32 + lane) * 16;
            ar = wmma_f16(a, *(const v16h*)(bLr + co), ar);
            az = wmma_f16(a, *(const v16h*)(bLz + co), az);
            an = wmma_f16(a, *(const v16h*)(wbn + co), an);
        }

        // prefetch next timestep's x_proj line for this lane's first row
        if (t + 1 < SEQ) {
            size_t nb = ((size_t)(bt * 16 + hoff) * SEQ + (t + 1)) * G3 + col;
            __builtin_prefetch(xp + nb, 0, 1);
        }

        #pragma unroll
        for (int j = 0; j < 8; ++j) {
            int m = j + hoff;                          // C layout row within tile
            int b = bt * 16 + m;
            size_t xbase = ((size_t)b * SEQ + t) * G3;
            float xr = (float)xp[xbase + col];
            float xz = (float)xp[xbase + H + col];
            float xn = (float)xp[xbase + 2 * H + col];
            float hold = (float)hcur[m * LDS_STRIDE + col];

            float r  = sigmoid_fast(xr + ar[j] + bh_r);
            float z  = sigmoid_fast(xz + az[j] + bh_z);
            float nn = tanh_fast(xn + r * (an[j] + bh_n));
            float hnew = (1.0f - z) * nn + z * hold;

            hnxt[m * LDS_STRIDE + col] = (_Float16)hnew;
            ann[((size_t)b * SEQ + t) * H + col] = hnew;
            if (t == SEQ - 1) hid[(size_t)b * H + col] = hnew;
        }
        __syncthreads();
    }
}

// ---------------------------------------------------------------------------
extern "C" void kernel_launch(void* const* d_in, const int* in_sizes, int n_in,
                              void* d_out, int out_size, void* d_ws, size_t ws_size,
                              hipStream_t stream) {
    const int*   inputs = (const int*)d_in[0];   // [B,S] int32
    const float* emb    = (const float*)d_in[1]; // [VOCAB,H]
    const float* W_ih   = (const float*)d_in[2]; // [3H,H]
    const float* W_hh   = (const float*)d_in[3]; // [3H,H]
    const float* b_ih   = (const float*)d_in[4]; // [3H]
    const float* b_hh   = (const float*)d_in[5]; // [3H]
    (void)in_sizes; (void)n_in; (void)out_size; (void)ws_size;

    // workspace layout (f16 halves): [W_ih swz | W_hh swz | x_proj]
    unsigned short* wih_swz = (unsigned short*)d_ws;               // 196608 halves
    unsigned short* whh_swz = wih_swz + (size_t)G3 * H;            // 196608 halves
    unsigned short* xproj   = whh_swz + (size_t)G3 * H;            // 65536*768 halves

    pack_weights<<<G3 * H / 256, 256, 0, stream>>>(W_ih, wih_swz);
    pack_weights<<<G3 * H / 256, 256, 0, stream>>>(W_hh, whh_swz);
    xproj_kernel<<<NROWS / 16, 256, 0, stream>>>(inputs, emb, wih_swz, b_ih, xproj);
    scan_kernel<<<BATCH / 16, 512, 0, stream>>>(whh_swz, xproj, b_hh, (float*)d_out);
}
